// BackgroundSuppression_72318659330122
// MI455X (gfx1250) — matrix operand order repaired and checked by reading, not compiled, not run on hardware
//
#include <hip/hip_runtime.h>
#include <hip/hip_bf16.h>
#include <cstdint>

typedef __attribute__((ext_vector_type(16))) _Float16 v16h;
typedef __attribute__((ext_vector_type(8)))  _Float16 v8h;
typedef __attribute__((ext_vector_type(8)))  float    v8f;

#define NB   16
#define NC   256
#define NH   80
#define NW   80
#define MID  64
#define HW   (NH*NW)          // 6400
#define KPAD 96               // 66 fuse channels padded to 96 (3 x 32 K-chunks)

// ---------------------------------------------------------------------------
// Kernel 0: convert weights to f16 with WMMA-friendly layouts.
//   projW16  : [m][256]          (row-major, contiguous K)
//   fuse1W16 : [tap][m][96]      (c<66 real, c>=66 zero)
// ---------------------------------------------------------------------------
__global__ __launch_bounds__(256)
void prep_weights(const float* __restrict__ proj_w, const float* __restrict__ fuse1_w,
                  _Float16* __restrict__ projW16, _Float16* __restrict__ fuse1W16) {
    int i = blockIdx.x * blockDim.x + threadIdx.x;
    if (i < MID * NC) projW16[i] = (_Float16)proj_w[i];
    if (i < 9 * MID * KPAD) {
        int k   = i % KPAD;
        int m   = (i / KPAD) % MID;
        int tap = i / (KPAD * MID);
        float v = 0.f;
        if (k < MID + 2) v = fuse1_w[(m * (MID + 2) + k) * 9 + tap];   // OIHW
        fuse1W16[i] = (_Float16)v;
    }
}

// ---------------------------------------------------------------------------
// Kernel 1: edge map (mean |sobel|) and period map (mean sqrt(var_lh+var_hl+eps))
// One 16x16 pixel tile per block, loop over 256 channels staged in LDS.
// ---------------------------------------------------------------------------
__global__ __launch_bounds__(256)
void edge_period_kernel(const float* __restrict__ x,
                        float* __restrict__ edge, float* __restrict__ per) {
    __shared__ float xs[24][25];          // x tile, global [-4..19]
    __shared__ float lhs[22][23];         // Haar LH, global [-3..18]
    __shared__ float hls[22][23];         // Haar HL
    __shared__ float rlh[20][21];         // resized LH, global [-2..17]
    __shared__ float rhl[20][21];         // resized HL

    const int bx0 = blockIdx.x * 16, by0 = blockIdx.y * 16, b = blockIdx.z;
    const int t = threadIdx.x, tx = t & 15, ty = t >> 4;
    const float* xb = x + (size_t)b * NC * HW;
    const float scale = (float)(NH - 1) / (float)NH;   // 79/80, half-pixel resize

    float esum = 0.f, psum = 0.f;

    for (int c = 0; c < NC; ++c) {
        const float* xc = xb + (size_t)c * HW;
        __syncthreads();                               // protect prior iter reads
        // ---- stage x with halo 4, zero outside image (sobel zero-pad) ----
        for (int e = t; e < 24 * 24; e += 256) {
            int r = e / 24, cc = e % 24;
            int gy = by0 - 4 + r, gx = bx0 - 4 + cc;
            float v = 0.f;
            if (gy >= 0 && gy < NH && gx >= 0 && gx < NW) v = xc[gy * NW + gx];
            xs[r][cc] = v;
        }
        __syncthreads();
        // ---- Haar 2x2 (valid) on global [-3..18] (junk where out-of-range; never read) ----
        for (int e = t; e < 22 * 22; e += 256) {
            int r = e / 22, cc = e % 22;               // xs offset: +1
            float a00 = xs[r + 1][cc + 1], a01 = xs[r + 1][cc + 2];
            float a10 = xs[r + 2][cc + 1], a11 = xs[r + 2][cc + 2];
            lhs[r][cc] = 0.5f * (a00 - a01 + a10 - a11);
            hls[r][cc] = 0.5f * (a00 + a01 - a10 - a11);
        }
        __syncthreads();
        // ---- bilinear resize (H-1 -> H, half-pixel, edge-clamped); 0 outside image ----
        for (int e = t; e < 20 * 20; e += 256) {
            int r = e / 20, cc = e % 20;
            int gy = by0 - 2 + r, gx = bx0 - 2 + cc;
            float vl = 0.f, vh = 0.f;
            if (gy >= 0 && gy < NH && gx >= 0 && gx < NW) {
                float uy = fminf(fmaxf((gy + 0.5f) * scale - 0.5f, 0.f), (float)(NH - 2));
                float ux = fminf(fmaxf((gx + 0.5f) * scale - 0.5f, 0.f), (float)(NW - 2));
                int iy0 = (int)uy; float fy = uy - iy0; int iy1 = min(iy0 + 1, NH - 2);
                int ix0 = (int)ux; float fx = ux - ix0; int ix1 = min(ix0 + 1, NW - 2);
                int ly0 = iy0 - (by0 - 3), ly1 = iy1 - (by0 - 3);
                int lx0 = ix0 - (bx0 - 3), lx1 = ix1 - (bx0 - 3);
                vl = (1.f - fy) * ((1.f - fx) * lhs[ly0][lx0] + fx * lhs[ly0][lx1])
                   +        fy  * ((1.f - fx) * lhs[ly1][lx0] + fx * lhs[ly1][lx1]);
                vh = (1.f - fy) * ((1.f - fx) * hls[ly0][lx0] + fx * hls[ly0][lx1])
                   +        fy  * ((1.f - fx) * hls[ly1][lx0] + fx * hls[ly1][lx1]);
            }
            rlh[r][cc] = vl;
            rhl[r][cc] = vh;
        }
        __syncthreads();
        // ---- per-pixel: sobel magnitude + 5x5 local variance of resized maps ----
        {
            float xm1m1 = xs[ty + 3][tx + 3], xm10 = xs[ty + 3][tx + 4], xm1p1 = xs[ty + 3][tx + 5];
            float x0m1  = xs[ty + 4][tx + 3],                            x0p1  = xs[ty + 4][tx + 5];
            float xp1m1 = xs[ty + 5][tx + 3], xp10 = xs[ty + 5][tx + 4], xp1p1 = xs[ty + 5][tx + 5];
            float gxs = (-xm1m1 + xm1p1 - 2.f * x0m1 + 2.f * x0p1 - xp1m1 + xp1p1) * 0.125f;
            float gys = (-xm1m1 - 2.f * xm10 - xm1p1 + xp1m1 + 2.f * xp10 + xp1p1) * 0.125f;
            esum += sqrtf(gxs * gxs + gys * gys);

            float s1 = 0.f, s2 = 0.f, s3 = 0.f, s4 = 0.f;
            #pragma unroll
            for (int dy = 0; dy < 5; ++dy)
                #pragma unroll
                for (int dx = 0; dx < 5; ++dx) {
                    float vl = rlh[ty + dy][tx + dx];
                    float vh = rhl[ty + dy][tx + dx];
                    s1 += vl; s2 += vl * vl; s3 += vh; s4 += vh * vh;
                }
            float ml = s1 * (1.f / 25.f), mh = s3 * (1.f / 25.f);
            float varl = fmaxf(s2 * (1.f / 25.f) - ml * ml, 0.f);
            float varh = fmaxf(s4 * (1.f / 25.f) - mh * mh, 0.f);
            psum += sqrtf(varl + varh + 1e-6f);
        }
    }
    size_t o = (size_t)b * HW + (size_t)(by0 + ty) * NW + bx0 + tx;
    edge[o] = esum * (1.f / (float)NC);
    per[o]  = psum * (1.f / (float)NC);
}

// ---------------------------------------------------------------------------
// Kernel 2: edge_density = edge / (avgpool5(edge) + 1e-6)
// ---------------------------------------------------------------------------
__global__ __launch_bounds__(256)
void edge_density_kernel(const float* __restrict__ edge, float* __restrict__ ed) {
    __shared__ float es[20][21];
    const int bx0 = blockIdx.x * 16, by0 = blockIdx.y * 16, b = blockIdx.z;
    const int t = threadIdx.x, tx = t & 15, ty = t >> 4;
    const float* eb = edge + (size_t)b * HW;
    for (int e = t; e < 20 * 20; e += 256) {
        int r = e / 20, cc = e % 20;
        int gy = by0 - 2 + r, gx = bx0 - 2 + cc;
        es[r][cc] = (gy >= 0 && gy < NH && gx >= 0 && gx < NW) ? eb[gy * NW + gx] : 0.f;
    }
    __syncthreads();
    float s = 0.f;
    #pragma unroll
    for (int dy = 0; dy < 5; ++dy)
        #pragma unroll
        for (int dx = 0; dx < 5; ++dx) s += es[ty + dy][tx + dx];
    float e0 = es[ty + 2][tx + 2];
    ed[(size_t)b * HW + (size_t)(by0 + ty) * NW + bx0 + tx] = e0 / (s * (1.f / 25.f) + 1e-6f);
}

// ---------------------------------------------------------------------------
// Kernel 3: feat = silu(bn1(proj_w @ x))  via WMMA f16 (M=64,K=256,N=B*HW)
// Block = 8 waves: 4 M-groups x 2 pixel-groups => 64x32 tile.
// Staging: one float4 (b128) per thread per K-chunk, transposed into LDS.
// feat stored pixel-major f16: feat[pix][64].
// ---------------------------------------------------------------------------
__global__ __launch_bounds__(256)
void proj_wmma(const float* __restrict__ x, const _Float16* __restrict__ Wp,
               const float* __restrict__ bn1g, const float* __restrict__ bn1b,
               const float* __restrict__ bn1m, const float* __restrict__ bn1v,
               _Float16* __restrict__ feat) {
    __shared__ __align__(32) _Float16 xs[32 * 48];     // [pix][k], pitch 48 halves
    __shared__ float sc1[MID], sh1[MID];

    const int t = threadIdx.x;
    const int pix0 = blockIdx.x * 32;                  // linear pixel over B*HW
    const int b = pix0 / HW;
    const float* xb = x + (size_t)b * NC * HW + (pix0 - b * HW);

    if (t < MID) {
        float inv = rsqrtf(bn1v[t] + 1e-5f);
        float s = bn1g[t] * inv;
        sc1[t] = s;
        sh1[t] = bn1b[t] - bn1m[t] * s;
    }

    const int wave = t >> 5, lane = t & 31;
    const int mbase = (wave & 3) * 16, pbase = (wave >> 2) * 16;
    const int scl = t >> 3;            // staging: channel 0..31
    const int sp4 = (t & 7) * 4;       // staging: pixel 0,4,...,28
    v8f acc = {};

    for (int kc = 0; kc < NC; kc += 32) {
        __syncthreads();
        // stage 32ch x 32pix chunk: one coalesced b128 load per thread, transpose to [pix][k]
        {
            const float4 v = *(const float4*)(xb + (size_t)(kc + scl) * HW + sp4);
            xs[(sp4 + 0) * 48 + scl] = (_Float16)v.x;
            xs[(sp4 + 1) * 48 + scl] = (_Float16)v.y;
            xs[(sp4 + 2) * 48 + scl] = (_Float16)v.z;
            xs[(sp4 + 3) * 48 + scl] = (_Float16)v.w;
        }
        __syncthreads();
        // A fragment (16x32 f16): two contiguous 16B runs per lane
        int m  = mbase + (lane & 15);
        int b0 = (lane < 16) ? 0 : 16;
        const _Float16* wrow = Wp + (size_t)m * NC + kc + b0;
        v16h a;
        ((v8h*)&a)[0] = *(const v8h*)(wrow);
        ((v8h*)&a)[1] = *(const v8h*)(wrow + 8);
        // B fragment (32x16 f16): 16 contiguous K halves per lane from LDS
        int kb = (lane < 16) ? 0 : 16;
        v16h bf = *(const v16h*)&xs[(pbase + (lane & 15)) * 48 + kb];
        acc = __builtin_amdgcn_wmma_f32_16x16x32_f16(false, a, false, bf,
                                                     (short)0, acc, false, false);
    }
    // epilogue: BN1 + SiLU, store f16 pixel-major (one 16B store per lane)
    const int n = lane & 15;
    const int off8 = (lane < 16) ? 0 : 8;
    const int gpix = pix0 + pbase + n;
    _Float16 outv[8];
    #pragma unroll
    for (int r = 0; r < 8; ++r) {
        int m = mbase + off8 + r;
        float z = acc[r] * sc1[m] + sh1[m];
        outv[r] = (_Float16)(z / (1.f + __expf(-z)));
    }
    *(v8h*)&feat[(size_t)gpix * MID + mbase + off8] = *(const v8h*)outv;
}

// ---------------------------------------------------------------------------
// Kernel 4: y = silu(bn2(conv3x3(combined))); w = sigmoid(fuse2 . y); out = x*w
// Block = 4 waves: 64 channels x 16 pixels (one row segment). 27 WMMAs.
// feat tile staged via CDNA5 async global->LDS b128 copies (ASYNCcnt).
// ---------------------------------------------------------------------------
__global__ __launch_bounds__(128)
void fuse_wmma(const float* __restrict__ x, const _Float16* __restrict__ feat,
               const float* __restrict__ ed, const float* __restrict__ per,
               const _Float16* __restrict__ Wf,
               const float* __restrict__ bn2g, const float* __restrict__ bn2b,
               const float* __restrict__ bn2m, const float* __restrict__ bn2v,
               const float* __restrict__ fuse2_w, float* __restrict__ out) {
    __shared__ __align__(32) _Float16 cs[3 * 18 * KPAD];   // [row3][col18][k96]
    __shared__ float sc2[MID], sh2[MID], w2[MID], wsum[16];

    const int t = threadIdx.x;
    const int x0 = blockIdx.x * 16, y = blockIdx.y, b = blockIdx.z;

    if (t < MID) {
        float inv = rsqrtf(bn2v[t] + 1e-5f);
        float s = bn2g[t] * inv;
        sc2[t] = s;
        sh2[t] = bn2b[t] - bn2m[t] * s;
        w2[t]  = fuse2_w[t];
    }
    if (t < 16) wsum[t] = 0.f;

    // ---- stage extra channels (edge_density | period | zero-pad), zero conv-pad ----
    for (int e = t; e < 3 * 18 * (KPAD - MID); e += 128) {
        int k   = MID + (e & 31);
        int pxi = e >> 5;
        int col = pxi % 18, row = pxi / 18;
        int gy = y + row - 1, gx = x0 + col - 1;
        _Float16 v = (_Float16)0.f;
        if (gy >= 0 && gy < NH && gx >= 0 && gx < NW) {
            size_t p = (size_t)b * HW + (size_t)gy * NW + gx;
            if (k == MID)          v = (_Float16)ed[p];
            else if (k == MID + 1) v = (_Float16)per[p];
        }
        cs[(row * 18 + col) * KPAD + k] = v;
    }
    // ---- stage feat channels: async global->LDS, 16B per copy (54 pix x 8 segs) ----
    for (int j = t; j < 3 * 18 * 8; j += 128) {
        int pxi = j >> 3, seg = j & 7;
        int col = pxi % 18, row = pxi / 18;
        int gy = y + row - 1, gx = x0 + col - 1;
        _Float16* dst = &cs[(row * 18 + col) * KPAD + seg * 8];
        if (gy >= 0 && gy < NH && gx >= 0 && gx < NW) {
            size_t p = (size_t)b * HW + (size_t)gy * NW + gx;
            const _Float16* g = feat + p * MID + seg * 8;
            uint32_t ldsb = (uint32_t)(uintptr_t)dst;    // LDS byte offset (low 32 bits)
            asm volatile("global_load_async_to_lds_b128 %0, %1, off"
                         :: "v"(ldsb), "v"(g) : "memory");
        } else {
            v8h z = {};
            *(v8h*)dst = z;                              // conv zero-pad
        }
    }
    asm volatile("s_wait_asynccnt 0x0" ::: "memory");
    __syncthreads();

    const int wave = t >> 5, lane = t & 31;
    const int mbase = wave * 16;
    v8f acc = {};

    for (int tap = 0; tap < 9; ++tap) {
        int dy = tap / 3, dx = tap % 3;
        #pragma unroll
        for (int j = 0; j < 3; ++j) {
            int kc = j * 32;
            int m  = mbase + (lane & 15);
            int b0 = (lane < 16) ? 0 : 16;
            const _Float16* wrow = Wf + ((size_t)tap * MID + m) * KPAD + kc + b0;
            v16h a;
            ((v8h*)&a)[0] = *(const v8h*)(wrow);
            ((v8h*)&a)[1] = *(const v8h*)(wrow + 8);
            int n  = lane & 15;
            int kb = kc + ((lane < 16) ? 0 : 16);
            v16h bf = *(const v16h*)&cs[((size_t)dy * 18 + n + dx) * KPAD + kb];
            acc = __builtin_amdgcn_wmma_f32_16x16x32_f16(false, a, false, bf,
                                                         (short)0, acc, false, false);
        }
    }

    // ---- epilogue: BN2+SiLU, 64->1 dot via LDS f32 atomics, sigmoid ----
    const int n = lane & 15;
    const int off8 = (lane < 16) ? 0 : 8;
    float partial = 0.f;
    #pragma unroll
    for (int r = 0; r < 8; ++r) {
        int m = mbase + off8 + r;
        float z = acc[r] * sc2[m] + sh2[m];
        float sv = z / (1.f + __expf(-z));
        partial += w2[m] * sv;
    }
    atomicAdd(&wsum[n], partial);
    __syncthreads();

    // ---- out = x * sigmoid(wsum): float4 (b128) loads/stores, coalesced ----
    const int col4 = (t & 3) * 4;
    const int crow = t >> 2;                              // 0..31
    float w4[4];
    #pragma unroll
    for (int i = 0; i < 4; ++i) w4[i] = 1.f / (1.f + __expf(-wsum[col4 + i]));
    size_t base = (size_t)b * NC * HW + (size_t)y * NW + x0 + col4;
    for (int c = crow; c < NC; c += 32) {
        size_t idx = base + (size_t)c * HW;
        float4 xv = *(const float4*)&x[idx];
        float4 ov = make_float4(xv.x * w4[0], xv.y * w4[1], xv.z * w4[2], xv.w * w4[3]);
        *(float4*)&out[idx] = ov;
    }
}

// ---------------------------------------------------------------------------
extern "C" void kernel_launch(void* const* d_in, const int* in_sizes, int n_in,
                              void* d_out, int out_size, void* d_ws, size_t ws_size,
                              hipStream_t stream) {
    const float* x       = (const float*)d_in[0];
    const float* proj_w  = (const float*)d_in[1];
    const float* bn1g    = (const float*)d_in[2];
    const float* bn1b    = (const float*)d_in[3];
    const float* bn1m    = (const float*)d_in[4];
    const float* bn1v    = (const float*)d_in[5];
    const float* fuse1_w = (const float*)d_in[6];
    const float* bn2g    = (const float*)d_in[7];
    const float* bn2b    = (const float*)d_in[8];
    const float* bn2m    = (const float*)d_in[9];
    const float* bn2v    = (const float*)d_in[10];
    const float* fuse2w  = (const float*)d_in[11];
    float* out = (float*)d_out;

    // workspace carve (all 256B aligned)
    char* w = (char*)d_ws;
    auto carve = [&](size_t bytes) { char* p = w; w += (bytes + 255) & ~(size_t)255; return p; };
    float*    edge     = (float*)   carve((size_t)NB * HW * sizeof(float));
    float*    per      = (float*)   carve((size_t)NB * HW * sizeof(float));
    float*    edens    = (float*)   carve((size_t)NB * HW * sizeof(float));
    _Float16* projW16  = (_Float16*)carve((size_t)MID * NC * sizeof(_Float16));
    _Float16* fuse1W16 = (_Float16*)carve((size_t)9 * MID * KPAD * sizeof(_Float16));
    _Float16* feat     = (_Float16*)carve((size_t)NB * HW * MID * sizeof(_Float16));

    prep_weights<<<dim3((9 * MID * KPAD + 255) / 256), 256, 0, stream>>>(
        proj_w, fuse1_w, projW16, fuse1W16);

    edge_period_kernel<<<dim3(NW / 16, NH / 16, NB), 256, 0, stream>>>(x, edge, per);

    edge_density_kernel<<<dim3(NW / 16, NH / 16, NB), 256, 0, stream>>>(edge, edens);

    proj_wmma<<<dim3((NB * HW) / 32), 256, 0, stream>>>(
        x, projW16, bn1g, bn1b, bn1m, bn1v, feat);

    fuse_wmma<<<dim3(NW / 16, NH, NB), 128, 0, stream>>>(
        x, feat, edens, per, fuse1W16, bn2g, bn2b, bn2m, bn2v, fuse2w, out);
}
